// CausalDownsamplingLRU_54898271977943
// MI455X (gfx1250) — compile-verified, matrix operand order checked by reading, not run on hardware
//
#include <hip/hip_runtime.h>
#include <hip/hip_bf16.h>

// ---------------------------------------------------------------------------
// CausalDownsamplingLRU for MI455X (gfx1250, wave32, WMMA)
//   phase A: bf16 pack of x and weights
//   phase B: projection GEMM  Bu = (x Bre^T | x Bim^T) * gamma   (WMMA bf16)
//            double-buffered LDS, async global->LDS copies (ASYNCcnt)
//   phase C: 3-phase chunked complex scan (lambda is time-invariant)
//   phase D: fused output GEMM y = [h_re|h_im|x_tail] x [Cre|-Cim|D]^T (WMMA)
// ---------------------------------------------------------------------------

typedef __bf16 bf16;
typedef __attribute__((ext_vector_type(16))) __bf16 v16bf;
typedef __attribute__((ext_vector_type(8)))  float  v8f;
typedef __attribute__((ext_vector_type(4)))  int    i4t;

#define LRU_BATCH 8
#define LRU_T     2048
#define LRU_IN    512
#define LRU_N     512
#define LRU_OUT   512
#define LRU_DS    1024
#define LRU_MROWS (LRU_BATCH * LRU_T)   // 16384 rows for projection GEMM
#define LRU_MOUT  (LRU_BATCH * LRU_DS)  // 8192 rows for output GEMM
#define LRU_CHUNK 128
#define LRU_NCHUNK (LRU_T / LRU_CHUNK)  // 16

#define BM 128
#define BN 128
#define BK 32
#define LDT 40   // padded LDS stride in halves (80 bytes, 16B aligned)

#if defined(__gfx1250__) && __has_builtin(__builtin_amdgcn_global_load_async_to_lds_b128)
#define LRU_ASYNC_LDS 1
#else
#define LRU_ASYNC_LDS 0
#endif

union FragBf { v16bf v; uint4 q[2]; };

#if LRU_ASYNC_LDS
typedef __attribute__((address_space(3))) i4t* lds_i4p;
__device__ __forceinline__ lds_i4p lru_to_lds(void* p) {
    return (lds_i4p)(__attribute__((address_space(3))) void*)p;
}
#endif

// 16-byte global -> LDS stage: async DMA path on gfx1250, sync fallback else.
__device__ __forceinline__ void stage16(const bf16* g, bf16* l) {
#if LRU_ASYNC_LDS
    __builtin_amdgcn_global_load_async_to_lds_b128(
        (i4t*)(void*)const_cast<bf16*>(g),   // global source (generic int4*)
        lru_to_lds(l),                       // LDS destination (AS3 int4*)
        0, 0);
#else
    *(uint4*)l = *(const uint4*)g;
#endif
}

__device__ __forceinline__ void wait_async_all() {
#if LRU_ASYNC_LDS
#if __has_builtin(__builtin_amdgcn_s_wait_asynccnt)
    __builtin_amdgcn_s_wait_asynccnt(0);
#else
    asm volatile("s_wait_asynccnt 0x0" ::: "memory");
#endif
#endif
}

// ---------------------------------------------------------------------------
// pack kernels (fp32 -> bf16)
// ---------------------------------------------------------------------------
__global__ void lru_convert_x(const float* __restrict__ x, bf16* __restrict__ xbf) {
    int gid = blockIdx.x * blockDim.x + threadIdx.x;   // over MROWS*IN
    xbf[gid] = (bf16)x[gid];
}

__global__ void lru_pack_wproj(const float* __restrict__ B_re,
                               const float* __restrict__ B_im,
                               bf16* __restrict__ wproj) {
    int gid = blockIdx.x * blockDim.x + threadIdx.x;   // over 512*512
    int n = gid >> 9, i = gid & 511;
    wproj[(size_t)n * 512 + i]         = (bf16)B_re[(size_t)n * 512 + i];
    wproj[(size_t)(512 + n) * 512 + i] = (bf16)B_im[(size_t)n * 512 + i];
}

__global__ void lru_pack_wout(const float* __restrict__ C_re,
                              const float* __restrict__ C_im,
                              const float* __restrict__ Dm,
                              bf16* __restrict__ wout) {
    int gid = blockIdx.x * blockDim.x + threadIdx.x;   // over 512*512
    int o = gid >> 9, k = gid & 511;
    size_t src = (size_t)o * 512 + k;
    size_t dst = (size_t)o * 1536;
    wout[dst + k]        = (bf16)C_re[src];
    wout[dst + 512 + k]  = (bf16)(-C_im[src]);
    wout[dst + 1024 + k] = (bf16)Dm[src];
}

// ---------------------------------------------------------------------------
// projection GEMM: [16384 x 512] x [1024 x 512]^T -> Bu_re / Bu_im (f32, *gamma)
// double-buffered LDS with async global->LDS staging
// ---------------------------------------------------------------------------
__global__ __launch_bounds__(256)
void lru_proj_gemm(const bf16* __restrict__ A,     // [MROWS, 512] bf16
                   const bf16* __restrict__ W,     // [1024, 512]  bf16
                   const float* __restrict__ gamma_log,
                   float* __restrict__ bu_re,
                   float* __restrict__ bu_im) {
    __shared__ __align__(16) bf16 lA[2][BM * LDT];
    __shared__ __align__(16) bf16 lB[2][BN * LDT];

    const int tid  = threadIdx.x;
    const int lane = tid & 31;
    const int wave = tid >> 5;
    const int wm   = wave & 3;        // 4 waves along M  -> 32 rows each
    const int wn   = wave >> 2;       // 2 waves along N  -> 64 cols each
    const int m0   = blockIdx.x * BM;
    const int n0   = blockIdx.y * BN;
    const int K    = 512;

    v8f acc[2][4];
#pragma unroll
    for (int i = 0; i < 2; ++i)
#pragma unroll
        for (int j = 0; j < 4; ++j) acc[i][j] = (v8f)0.0f;

    const int lrow = lane & 15;
    const int krow = (lane >> 4) << 3;    // 0 or 8 halves

    auto stage_tile = [&](int kk, int buf) {
#pragma unroll
        for (int c = tid; c < (BM * BK) / 8; c += 256) {
            int r  = c >> 2;
            int cc = (c & 3) << 3;
            stage16(&A[(size_t)(m0 + r) * K + kk + cc], &lA[buf][r * LDT + cc]);
            stage16(&W[(size_t)(n0 + r) * K + kk + cc], &lB[buf][r * LDT + cc]);
        }
    };

    stage_tile(0, 0);
    wait_async_all();
    __syncthreads();

    int buf = 0;
    for (int kk = 0; kk < K; kk += BK) {
        if (kk + BK < K) stage_tile(kk + BK, buf ^ 1);   // fill next buffer
        if (kk + 2 * BK < K) {   // gfx1250 global_prefetch_b8, 2 tiles ahead
            __builtin_prefetch(&A[(size_t)(m0 + (tid >> 1)) * K + kk + 2 * BK], 0, 3);
            __builtin_prefetch(&W[(size_t)(n0 + (tid >> 1)) * K + kk + 2 * BK], 0, 3);
        }

        FragBf af[2], bfr[4];
#pragma unroll
        for (int i = 0; i < 2; ++i) {
            int row = wm * 32 + i * 16 + lrow;
            af[i].q[0] = *(const uint4*)&lA[buf][row * LDT + krow];
            af[i].q[1] = *(const uint4*)&lA[buf][row * LDT + krow + 16];
        }
#pragma unroll
        for (int j = 0; j < 4; ++j) {
            int row = wn * 64 + j * 16 + lrow;
            bfr[j].q[0] = *(const uint4*)&lB[buf][row * LDT + krow];
            bfr[j].q[1] = *(const uint4*)&lB[buf][row * LDT + krow + 16];
        }
#pragma unroll
        for (int i = 0; i < 2; ++i)
#pragma unroll
            for (int j = 0; j < 4; ++j)
                acc[i][j] = __builtin_amdgcn_wmma_f32_16x16x32_bf16(
                    false, af[i].v, false, bfr[j].v, (short)0, acc[i][j],
                    false, false);

        wait_async_all();     // next buffer fully in LDS (per-wave ASYNCcnt)
        __syncthreads();      // all waves' copies + this buffer's reads done
        buf ^= 1;
    }

    // epilogue: scale by gamma, split into re / im planes
#pragma unroll
    for (int j = 0; j < 4; ++j) {
        int n  = n0 + wn * 64 + j * 16 + lrow;
        int nn = n & 511;
        float g = expf(gamma_log[nn]);
        float* dst = (n < 512) ? bu_re : bu_im;
#pragma unroll
        for (int i = 0; i < 2; ++i) {
            int mb = m0 + wm * 32 + i * 16 + krow;
#pragma unroll
            for (int v = 0; v < 8; ++v)
                dst[(size_t)(mb + v) * 512 + nn] = acc[i][j][v] * g;
        }
    }
}

// ---------------------------------------------------------------------------
// scan phase 1: chunk-local scans, in place over Bu; emit chunk carries
// ---------------------------------------------------------------------------
__global__ void lru_scan_phase1(const float* __restrict__ nu_log,
                                const float* __restrict__ theta_log,
                                float* __restrict__ bu_re,
                                float* __restrict__ bu_im,
                                float* __restrict__ carry_re,
                                float* __restrict__ carry_im) {
    int gid = blockIdx.x * blockDim.x + threadIdx.x;  // BATCH*NCHUNK*512
    int n = gid & 511;
    int c = (gid >> 9) & (LRU_NCHUNK - 1);
    int b = gid >> 13;

    float amp = expf(-expf(nu_log[n]));
    float ang = expf(theta_log[n]);
    float lr = amp * cosf(ang), li = amp * sinf(ang);

    float hr = 0.f, hi = 0.f;
    size_t base = ((size_t)b * LRU_T + (size_t)c * LRU_CHUNK) * 512 + n;
    for (int j = 0; j < LRU_CHUNK; ++j) {
        size_t idx = base + (size_t)j * 512;
        float br = bu_re[idx], bi = bu_im[idx];
        float nr = lr * hr - li * hi + br;
        float ni = lr * hi + li * hr + bi;
        hr = nr; hi = ni;
        bu_re[idx] = hr; bu_im[idx] = hi;
    }
    carry_re[gid] = hr;
    carry_im[gid] = hi;
}

// ---------------------------------------------------------------------------
// scan phase 2: sequential combine of 16 chunk carries per (b, n)
// ---------------------------------------------------------------------------
__global__ void lru_scan_phase2(const float* __restrict__ nu_log,
                                const float* __restrict__ theta_log,
                                const float* __restrict__ carry_re,
                                const float* __restrict__ carry_im,
                                float* __restrict__ pre_re,
                                float* __restrict__ pre_im) {
    int gid = blockIdx.x * blockDim.x + threadIdx.x;  // BATCH*512
    int n = gid & 511, b = gid >> 9;

    float amp = expf(-expf(nu_log[n]));
    float ang = expf(theta_log[n]);
    float lr = amp * cosf(ang), li = amp * sinf(ang);

    // lambda^128 via 7 complex squarings
    float pr = lr, pi = li;
#pragma unroll
    for (int s = 0; s < 7; ++s) {
        float t = pr * pr - pi * pi;
        pi = 2.f * pr * pi;
        pr = t;
    }

    float hr = 0.f, hi = 0.f;
    for (int c = 0; c < LRU_NCHUNK; ++c) {
        size_t idx = ((size_t)b * LRU_NCHUNK + c) * 512 + n;
        float nr = pr * hr - pi * hi + carry_re[idx];
        float ni = pr * hi + pi * hr + carry_im[idx];
        hr = nr; hi = ni;
        pre_re[idx] = hr; pre_im[idx] = hi;
    }
}

// ---------------------------------------------------------------------------
// scan phase 3: add lambda^{j+1} * prefix for output chunks 8..15, emit bf16 h
// ---------------------------------------------------------------------------
__global__ void lru_scan_phase3(const float* __restrict__ nu_log,
                                const float* __restrict__ theta_log,
                                const float* __restrict__ bu_re,
                                const float* __restrict__ bu_im,
                                const float* __restrict__ pre_re,
                                const float* __restrict__ pre_im,
                                bf16* __restrict__ hre,
                                bf16* __restrict__ him) {
    int gid = blockIdx.x * blockDim.x + threadIdx.x;  // BATCH*8*512
    int n = gid & 511;
    int c = 8 + ((gid >> 9) & 7);
    int b = gid >> 12;

    float amp = expf(-expf(nu_log[n]));
    float ang = expf(theta_log[n]);
    float lr = amp * cosf(ang), li = amp * sinf(ang);

    size_t pidx = ((size_t)b * LRU_NCHUNK + (c - 1)) * 512 + n;
    float Pr = pre_re[pidx], Pi = pre_im[pidx];

    float pr = lr, pi = li;  // lambda^(j+1), j starting at 0
    size_t base_in  = ((size_t)b * LRU_T + (size_t)c * LRU_CHUNK) * 512 + n;
    size_t base_out = ((size_t)b * LRU_DS + (size_t)(c - 8) * LRU_CHUNK) * 512 + n;
    for (int j = 0; j < LRU_CHUNK; ++j) {
        float hr = bu_re[base_in + (size_t)j * 512] + (pr * Pr - pi * Pi);
        float hi = bu_im[base_in + (size_t)j * 512] + (pr * Pi + pi * Pr);
        hre[base_out + (size_t)j * 512] = (bf16)hr;
        him[base_out + (size_t)j * 512] = (bf16)hi;
        float t = pr * lr - pi * li;
        pi = pr * li + pi * lr;
        pr = t;
    }
}

// ---------------------------------------------------------------------------
// output GEMM: [8192 x 1536] (h_re|h_im|x_tail) x [512 x 1536]^T -> y (f32)
// double-buffered LDS with async global->LDS staging, segmented A source
// ---------------------------------------------------------------------------
__global__ __launch_bounds__(256)
void lru_out_gemm(const bf16* __restrict__ hre,
                  const bf16* __restrict__ him,
                  const bf16* __restrict__ xbf,
                  const bf16* __restrict__ W,    // [512, 1536] bf16
                  float* __restrict__ y) {
    __shared__ __align__(16) bf16 lA[2][BM * LDT];
    __shared__ __align__(16) bf16 lB[2][BN * LDT];

    const int tid  = threadIdx.x;
    const int lane = tid & 31;
    const int wave = tid >> 5;
    const int wm   = wave & 3;
    const int wn   = wave >> 2;
    const int m0   = blockIdx.x * BM;
    const int n0   = blockIdx.y * BN;
    const int K    = 1536;

    v8f acc[2][4];
#pragma unroll
    for (int i = 0; i < 2; ++i)
#pragma unroll
        for (int j = 0; j < 4; ++j) acc[i][j] = (v8f)0.0f;

    const int lrow = lane & 15;
    const int krow = (lane >> 4) << 3;

    auto stage_tile = [&](int kk, int bufi) {
#pragma unroll
        for (int c = tid; c < (BM * BK) / 8; c += 256) {
            int r  = c >> 2;
            int cc = (c & 3) << 3;
            int mrow = m0 + r;
            int klocal = kk + cc;
            int seg = klocal >> 9;      // 0: h_re, 1: h_im, 2: x tail
            int ks  = klocal & 511;
            const bf16* src;
            size_t rowoff;
            if (seg == 0)      { src = hre; rowoff = (size_t)mrow * 512; }
            else if (seg == 1) { src = him; rowoff = (size_t)mrow * 512; }
            else {
                int b = mrow >> 10;          // / DS
                int t = mrow & 1023;         // % DS
                src = xbf;
                rowoff = ((size_t)b * LRU_T + (LRU_T - LRU_DS) + t) * 512;
            }
            stage16(&src[rowoff + ks], &lA[bufi][r * LDT + cc]);
            stage16(&W[(size_t)(n0 + r) * K + kk + cc], &lB[bufi][r * LDT + cc]);
        }
    };

    stage_tile(0, 0);
    wait_async_all();
    __syncthreads();

    int buf = 0;
    for (int kk = 0; kk < K; kk += BK) {
        if (kk + BK < K) stage_tile(kk + BK, buf ^ 1);
        if (kk + 2 * BK < K)
            __builtin_prefetch(&W[(size_t)(n0 + (tid >> 1)) * K + kk + 2 * BK], 0, 3);

        FragBf af[2], bfr[4];
#pragma unroll
        for (int i = 0; i < 2; ++i) {
            int row = wm * 32 + i * 16 + lrow;
            af[i].q[0] = *(const uint4*)&lA[buf][row * LDT + krow];
            af[i].q[1] = *(const uint4*)&lA[buf][row * LDT + krow + 16];
        }
#pragma unroll
        for (int j = 0; j < 4; ++j) {
            int row = wn * 64 + j * 16 + lrow;
            bfr[j].q[0] = *(const uint4*)&lB[buf][row * LDT + krow];
            bfr[j].q[1] = *(const uint4*)&lB[buf][row * LDT + krow + 16];
        }
#pragma unroll
        for (int i = 0; i < 2; ++i)
#pragma unroll
            for (int j = 0; j < 4; ++j)
                acc[i][j] = __builtin_amdgcn_wmma_f32_16x16x32_bf16(
                    false, af[i].v, false, bfr[j].v, (short)0, acc[i][j],
                    false, false);

        wait_async_all();
        __syncthreads();
        buf ^= 1;
    }

#pragma unroll
    for (int j = 0; j < 4; ++j) {
        int n = n0 + wn * 64 + j * 16 + lrow;
#pragma unroll
        for (int i = 0; i < 2; ++i) {
            int mb = m0 + wm * 32 + i * 16 + krow;
#pragma unroll
            for (int v = 0; v < 8; ++v)
                y[(size_t)(mb + v) * LRU_OUT + n] = acc[i][j][v];
        }
    }
}

// ---------------------------------------------------------------------------
// launch
// ---------------------------------------------------------------------------
extern "C" void kernel_launch(void* const* d_in, const int* in_sizes, int n_in,
                              void* d_out, int out_size, void* d_ws, size_t ws_size,
                              hipStream_t stream) {
    const float* x         = (const float*)d_in[0];
    const float* nu_log    = (const float*)d_in[1];
    const float* theta_log = (const float*)d_in[2];
    const float* gamma_log = (const float*)d_in[3];
    const float* B_re      = (const float*)d_in[4];
    const float* B_im      = (const float*)d_in[5];
    const float* C_re      = (const float*)d_in[6];
    const float* C_im      = (const float*)d_in[7];
    const float* Dm        = (const float*)d_in[8];
    float* y = (float*)d_out;

    char* p = (char*)d_ws;
    bf16* xbf      = (bf16*)p;  p += (size_t)LRU_MROWS * LRU_IN * 2;
    bf16* wproj    = (bf16*)p;  p += (size_t)1024 * 512 * 2;
    bf16* wout     = (bf16*)p;  p += (size_t)512 * 1536 * 2;
    float* bu_re   = (float*)p; p += (size_t)LRU_MROWS * 512 * 4;
    float* bu_im   = (float*)p; p += (size_t)LRU_MROWS * 512 * 4;
    float* carry_re= (float*)p; p += (size_t)LRU_BATCH * LRU_NCHUNK * 512 * 4;
    float* carry_im= (float*)p; p += (size_t)LRU_BATCH * LRU_NCHUNK * 512 * 4;
    float* pre_re  = (float*)p; p += (size_t)LRU_BATCH * LRU_NCHUNK * 512 * 4;
    float* pre_im  = (float*)p; p += (size_t)LRU_BATCH * LRU_NCHUNK * 512 * 4;
    bf16* hre      = (bf16*)p;  p += (size_t)LRU_MOUT * 512 * 2;
    bf16* him      = (bf16*)p;  p += (size_t)LRU_MOUT * 512 * 2;

    lru_convert_x<<<(LRU_MROWS * LRU_IN) / 256, 256, 0, stream>>>(x, xbf);
    lru_pack_wproj<<<(512 * 512) / 256, 256, 0, stream>>>(B_re, B_im, wproj);
    lru_pack_wout<<<(512 * 512) / 256, 256, 0, stream>>>(C_re, C_im, Dm, wout);

    lru_proj_gemm<<<dim3(LRU_MROWS / BM, 1024 / BN), 256, 0, stream>>>(
        xbf, wproj, gamma_log, bu_re, bu_im);

    lru_scan_phase1<<<(LRU_BATCH * LRU_NCHUNK * 512) / 256, 256, 0, stream>>>(
        nu_log, theta_log, bu_re, bu_im, carry_re, carry_im);
    lru_scan_phase2<<<(LRU_BATCH * 512) / 256, 256, 0, stream>>>(
        nu_log, theta_log, carry_re, carry_im, pre_re, pre_im);
    lru_scan_phase3<<<(LRU_BATCH * 8 * 512) / 256, 256, 0, stream>>>(
        nu_log, theta_log, bu_re, bu_im, pre_re, pre_im, hre, him);

    lru_out_gemm<<<dim3(LRU_MOUT / BM, LRU_OUT / BN), 256, 0, stream>>>(
        hre, him, xbf, wout, y);
}
